// HierarchicalSelfAttention_30193620090907
// MI455X (gfx1250) — matrix operand order repaired and checked by reading, not compiled, not run on hardware
//
#include <hip/hip_runtime.h>
#include <math.h>

typedef __attribute__((ext_vector_type(16))) _Float16 v16h;
typedef __attribute__((ext_vector_type(8)))  _Float16 v8h;
typedef __attribute__((ext_vector_type(8)))  float    v8f;
typedef int v4i __attribute__((vector_size(16)));

#if defined(__has_builtin)
#  if __has_builtin(__builtin_amdgcn_global_load_async_to_lds_b128)
#    define HAVE_ASYNC_LDS 1
#  endif
#endif
#ifndef HAVE_ASYNC_LDS
#  define HAVE_ASYNC_LDS 0
#endif

// ---------- helpers ----------
__device__ __forceinline__ v16h cat16(v8h a, v8h b) {
  return __builtin_shufflevector(a, b, 0,1,2,3,4,5,6,7,8,9,10,11,12,13,14,15);
}
__device__ __forceinline__ v8f wmma_f16(v16h a, v16h b, v8f c) {
  return __builtin_amdgcn_wmma_f32_16x16x32_f16(false, a, false, b, (short)0, c, false, false);
}
__device__ __forceinline__ float wred_sum(float v) {
  #pragma unroll
  for (int o = 16; o > 0; o >>= 1) v += __shfl_xor(v, o, 32);
  return v;
}
__device__ __forceinline__ float wred_max(float v) {
  #pragma unroll
  for (int o = 16; o > 0; o >>= 1) v = fmaxf(v, __shfl_xor(v, o, 32));
  return v;
}

// Problem constants (B=2, S=8192, D=512, H=8, hd=64, G=64, L=128)
#define MROWS 16384   // B*S
#define DDIM  512
#define NHEAD 8
#define HD    64
#define NGRP  64
#define LTOK  128
#define KT32  16      // DDIM/32

// ---------- 1. LayerNorm (f32 in -> f16 out) ----------
__global__ __launch_bounds__(256) void ln_kernel(
    const float* __restrict__ x, const float* __restrict__ gam,
    const float* __restrict__ bet, _Float16* __restrict__ xn) {
  int wave = threadIdx.x >> 5, lane = threadIdx.x & 31;
  int row = blockIdx.x * 8 + wave;
  const float* xr = x + (size_t)row * DDIM;
  float v[16], s = 0.f, s2 = 0.f;
  #pragma unroll
  for (int i = 0; i < 16; ++i) {
    float t = xr[lane + i * 32];
    v[i] = t; s += t; s2 += t * t;
  }
  s = wred_sum(s); s2 = wred_sum(s2);
  float mu = s * (1.f / DDIM);
  float var = s2 * (1.f / DDIM) - mu * mu;
  float r = rsqrtf(var + 1e-5f);
  _Float16* o = xn + (size_t)row * DDIM;
  #pragma unroll
  for (int i = 0; i < 16; ++i) {
    int c = lane + i * 32;
    o[c] = (_Float16)(((v[i] - mu) * r) * gam[c] + bet[c]);
  }
}

// ---------- 2. Weight prepack: W[n][k] f32 -> fragment-major f16 B-layout ----------
// Bp[((nt*KT32+kt)*32 + lane)*16 + h] = W[nt*16+(lane&15)][kt*32+(lane>>4)*16+h]
__global__ __launch_bounds__(256) void prepack(const float* __restrict__ W,
                                               _Float16* __restrict__ Bp) {
  int idx = blockIdx.x * 256 + threadIdx.x;   // 0..262143
  int h  = idx & 15;
  int L  = (idx >> 4) & 31;
  int kt = (idx >> 9) & 15;
  int nt = idx >> 13;
  int n = nt * 16 + (L & 15);
  int k = kt * 32 + (L >> 4) * 16 + h;
  Bp[idx] = (_Float16)W[(size_t)n * DDIM + k];
}

// ---------- 3. GEMM: C[16384,512] = A[16384,512] * Bpacked ----------
// Each wave: one 16-row m-tile x four 16-col n-tiles. 8 waves/block.
template <bool RES>
__global__ __launch_bounds__(256) void gemm512(
    const _Float16* __restrict__ A, const _Float16* __restrict__ Bp,
    const float* __restrict__ resid, _Float16* __restrict__ outH,
    float* __restrict__ outF) {
  int wave = threadIdx.x >> 5, lane = threadIdx.x & 31;
  int ln = lane & 15, lh = lane >> 4;
  int mt = blockIdx.x * 8 + wave;
  int nt0 = blockIdx.y * 4;
  v8f acc[4];
  #pragma unroll
  for (int j = 0; j < 4; ++j)
    #pragma unroll
    for (int v = 0; v < 8; ++v) acc[j][v] = 0.f;
  const _Float16* arow = A + (size_t)(mt * 16 + ln) * DDIM;
  for (int kt = 0; kt < KT32; ++kt) {
    v16h a = cat16(*(const v8h*)(arow + kt * 32 + lh * 16),
                   *(const v8h*)(arow + kt * 32 + lh * 16 + 8));
    #pragma unroll
    for (int j = 0; j < 4; ++j) {
      const _Float16* bp = Bp + ((size_t)((nt0 + j) * KT32 + kt) * 32 + lane) * 16;
      v16h b = cat16(*(const v8h*)bp, *(const v8h*)(bp + 8));
      acc[j] = wmma_f16(a, b, acc[j]);
    }
  }
  #pragma unroll
  for (int j = 0; j < 4; ++j) {
    int col = (nt0 + j) * 16 + ln;
    #pragma unroll
    for (int v = 0; v < 8; ++v) {
      size_t o = (size_t)(mt * 16 + v + 8 * lh) * DDIM + col;
      if (RES) outF[o] = resid[o] + acc[j][v];
      else     outH[o] = (_Float16)acc[j][v];
    }
  }
}

// ---------- 4. L2-normalize each 64-dim head vector, scale by sqrt(hd)=8 ----------
__global__ __launch_bounds__(256) void qknorm(_Float16* __restrict__ X) {
  int wave = threadIdx.x >> 5, lane = threadIdx.x & 31;
  size_t vec = (size_t)blockIdx.x * 8 + wave;       // B*S*H vectors of 64
  _Float16* p = X + vec * HD + lane * 2;
  float a = (float)p[0], b = (float)p[1];
  float s = wred_sum(a * a + b * b);
  float sc = 8.0f / fmaxf(sqrtf(s), 1e-12f);
  p[0] = (_Float16)(a * sc);
  p[1] = (_Float16)(b * sc);
}

// ---------- 5. Per-(b,h,g) means: rho_q, rho_k, mu_v [B,H,G,64] f32 ----------
__global__ __launch_bounds__(256) void group_stats(
    const _Float16* __restrict__ Q, const _Float16* __restrict__ K,
    const _Float16* __restrict__ V, float* __restrict__ rq,
    float* __restrict__ rk, float* __restrict__ mv) {
  int blk = blockIdx.x;            // ((b*H)+h)*G + g
  int t = threadIdx.x;
  if (t >= 192) return;
  int which = t >> 6, d = t & 63;
  int g = blk & 63, hh = (blk >> 6) & 7, b = blk >> 9;
  const _Float16* X = (which == 0) ? Q : (which == 1) ? K : V;
  size_t base = ((size_t)b * 8192 + g * LTOK) * DDIM + hh * HD + d;
  float s = 0.f;
  for (int l = 0; l < LTOK; ++l) s += (float)X[base + (size_t)l * DDIM];
  s *= (1.f / LTOK);
  float* O = (which == 0) ? rq : (which == 1) ? rk : mv;
  O[(size_t)blk * 64 + d] = s;
}

// ---------- 6. Sibling interaction per (b,h): mu_w[G], inter[G,64] ----------
__global__ __launch_bounds__(256) void sibling(
    const float* __restrict__ rq, const float* __restrict__ rk,
    const float* __restrict__ mv, float* __restrict__ mu_w,
    float* __restrict__ inter) {
  __shared__ float sq[64 * 64], sk[64 * 64], lw[64 * 64], ld[64];
  int bh = blockIdx.x, t = threadIdx.x;
  const float* RQ = rq + (size_t)bh * 4096;
  const float* RK = rk + (size_t)bh * 4096;
  for (int i = t; i < 4096; i += 256) { sq[i] = RQ[i]; sk[i] = RK[i]; }
  __syncthreads();
  for (int p = t; p < 4096; p += 256) {
    int g = p >> 6, s = p & 63;
    float dot = 0.f;
    for (int d = 0; d < 64; ++d) dot += sq[g * 64 + d] * sk[s * 64 + d];
    // logw = log(L) + dots/hd - hd ; diag = -inf
    float w = 4.852030263919617f + dot * (1.f / 64.f) - 64.f;
    lw[p] = (g == s) ? -INFINITY : w;
  }
  __syncthreads();
  if (t < 64) {
    float m = -10000.f;   // log_alpha term included in the max
    for (int s = 0; s < 64; ++s) m = fmaxf(m, lw[t * 64 + s]);
    float sum = __expf(-10000.f - m);
    for (int s = 0; s < 64; ++s)
      if (s != t) sum += __expf(lw[t * 64 + s] - m);
    float dnm = m + __logf(sum);
    ld[t] = dnm;
    mu_w[(size_t)bh * 64 + t] = __expf(-10000.f - dnm);
  }
  __syncthreads();
  for (int p = t; p < 4096; p += 256) {
    int g = p >> 6, s = p & 63;
    lw[p] = (g == s) ? 0.f : __expf(lw[p] - ld[g]);
  }
  __syncthreads();
  const float* MV = mv + (size_t)bh * 4096;
  for (int p = t; p < 4096; p += 256) {
    int g = p >> 6, d = p & 63;
    float s = 0.f;
    for (int s2 = 0; s2 < 64; ++s2) s += lw[g * 64 + s2] * MV[s2 * 64 + d];
    inter[(size_t)bh * 4096 + p] = s;
  }
}

// ---------- 7. Intra-block attention + combine ----------
// One block per (b,h,g); 8 waves, each wave owns 16 query rows. The V tile
// (128x64 f16 = 16KB) is staged into LDS once per workgroup via the CDNA5
// async-to-LDS path; score strips stay wave-local (no barrier for softmax).
__global__ __launch_bounds__(256) void intra_attn(
    const _Float16* __restrict__ Q, const _Float16* __restrict__ K,
    const _Float16* __restrict__ V, const float* __restrict__ mu_w,
    const float* __restrict__ inter, _Float16* __restrict__ O) {
  __shared__ __align__(16) unsigned char smem[128 * 128 * 4 + 128 * 128 * 2 + 128 * 64 * 2];
  float*    sc   = (float*)smem;
  _Float16* at   = (_Float16*)(smem + 128 * 128 * 4);
  _Float16* vraw = (_Float16*)(smem + 128 * 128 * 4 + 128 * 128 * 2);
  int blk = blockIdx.x;
  int g = blk & 63, hh = (blk >> 6) & 7, b = blk >> 9;
  int wave = threadIdx.x >> 5, lane = threadIdx.x & 31;
  int ln = lane & 15, lh = lane >> 4;
  size_t tokbase = (size_t)b * 8192 + g * LTOK;
  int colbase = hh * HD;

  // stage V tile [128 tokens x 64 dims] into LDS
  {
    const _Float16* gsrc = V + tokbase * DDIM + colbase;
#if HAVE_ASYNC_LDS
    #pragma unroll
    for (int it = 0; it < 4; ++it) {
      int idx8 = (threadIdx.x + it * 256) * 8;   // half-index, 8 halves per op
      int tok = idx8 >> 6, dcol = idx8 & 63;
      __builtin_amdgcn_global_load_async_to_lds_b128(
          (__attribute__((address_space(1))) v4i*)(void*)(gsrc + (size_t)tok * DDIM + dcol),
          (__attribute__((address_space(3))) v4i*)(void*)(vraw + idx8),
          0, 0);
    }
#if __has_builtin(__builtin_amdgcn_s_wait_asynccnt)
    __builtin_amdgcn_s_wait_asynccnt(0);
#else
    asm volatile("s_wait_asynccnt 0x0" ::: "memory");
#endif
#else
    #pragma unroll
    for (int it = 0; it < 4; ++it) {
      int idx8 = (threadIdx.x + it * 256) * 8;
      int tok = idx8 >> 6, dcol = idx8 & 63;
      *(v8h*)(vraw + idx8) = *(const v8h*)(gsrc + (size_t)tok * DDIM + dcol);
    }
#endif
  }
  __syncthreads();

  // phase 1: scores = Q Kt / 8  (M=16 rows per wave, N=128 keys, K=64 dims)
  v8f acc[8];
  #pragma unroll
  for (int j = 0; j < 8; ++j)
    #pragma unroll
    for (int v = 0; v < 8; ++v) acc[j][v] = 0.f;
  const _Float16* qrow = Q + (tokbase + wave * 16 + ln) * DDIM + colbase;
  #pragma unroll
  for (int kt = 0; kt < 2; ++kt) {
    v16h a = cat16(*(const v8h*)(qrow + kt * 32 + lh * 16),
                   *(const v8h*)(qrow + kt * 32 + lh * 16 + 8));
    #pragma unroll
    for (int j = 0; j < 8; ++j) {
      // B = K^T: lane = key token, contiguous dims -> free transpose
      const _Float16* kp = K + (tokbase + j * 16 + ln) * DDIM + colbase + kt * 32 + lh * 16;
      v16h bb = cat16(*(const v8h*)kp, *(const v8h*)(kp + 8));
      acc[j] = wmma_f16(a, bb, acc[j]);
    }
  }
  #pragma unroll
  for (int j = 0; j < 8; ++j)
    #pragma unroll
    for (int v = 0; v < 8; ++v)
      sc[(wave * 16 + v + 8 * lh) * 128 + j * 16 + ln] = acc[j][v] * 0.125f;

  // phase 2: softmax over each of the wave's 16 rows (wave-local strip)
  for (int r = 0; r < 16; ++r) {
    float* row = sc + (wave * 16 + r) * 128;
    float v0 = row[lane], v1 = row[lane + 32], v2 = row[lane + 64], v3 = row[lane + 96];
    float m = wred_max(fmaxf(fmaxf(v0, v1), fmaxf(v2, v3)));
    v0 = __expf(v0 - m); v1 = __expf(v1 - m);
    v2 = __expf(v2 - m); v3 = __expf(v3 - m);
    float s = wred_sum(v0 + v1 + v2 + v3);
    float inv = 1.f / s;
    _Float16* arow = at + (wave * 16 + r) * 128;
    arow[lane]      = (_Float16)(v0 * inv);
    arow[lane + 32] = (_Float16)(v1 * inv);
    arow[lane + 64] = (_Float16)(v2 * inv);
    arow[lane + 96] = (_Float16)(v3 * inv);
  }

  // phase 3: intra = attn @ V  (M=16, N=64 dims, K=128 keys), V from LDS
  v8f oac[4];
  #pragma unroll
  for (int j = 0; j < 4; ++j)
    #pragma unroll
    for (int v = 0; v < 8; ++v) oac[j][v] = 0.f;
  const _Float16* arow2 = at + (wave * 16 + ln) * 128;
  #pragma unroll
  for (int kt = 0; kt < 4; ++kt) {
    v16h a = cat16(*(const v8h*)(arow2 + kt * 32 + lh * 16),
                   *(const v8h*)(arow2 + kt * 32 + lh * 16 + 8));
    #pragma unroll
    for (int j = 0; j < 4; ++j) {
      int d = j * 16 + ln;
      v16h bb;
      #pragma unroll
      for (int h2 = 0; h2 < 16; ++h2)
        bb[h2] = vraw[(kt * 32 + lh * 16 + h2) * HD + d];
      oac[j] = wmma_f16(a, bb, oac[j]);
    }
  }

  // epilogue: out = mu_w * intra + inter (broadcast over rows)
  float mw = mu_w[(size_t)(b * NHEAD + hh) * NGRP + g];
  const float* itr = inter + ((size_t)(b * NHEAD + hh) * NGRP + g) * 64;
  #pragma unroll
  for (int j = 0; j < 4; ++j) {
    int d = j * 16 + ln;
    float ib = itr[d];
    #pragma unroll
    for (int v = 0; v < 8; ++v) {
      int row = wave * 16 + v + 8 * lh;
      O[(tokbase + row) * DDIM + colbase + d] = (_Float16)(mw * oac[j][v] + ib);
    }
  }
}

// ---------- host ----------
extern "C" void kernel_launch(void* const* d_in, const int* in_sizes, int n_in,
                              void* d_out, int out_size, void* d_ws, size_t ws_size,
                              hipStream_t stream) {
  (void)in_sizes; (void)n_in; (void)out_size; (void)ws_size;
  const float* x     = (const float*)d_in[0];
  const float* Wq    = (const float*)d_in[1];
  const float* Wk    = (const float*)d_in[2];
  const float* Wv    = (const float*)d_in[3];
  const float* Wo    = (const float*)d_in[4];
  const float* gamma = (const float*)d_in[5];
  const float* beta  = (const float*)d_in[6];

  char* ws = (char*)d_ws;
  size_t off = 0;
  auto alloc = [&](size_t bytes) -> void* {
    void* p = ws + off;
    off = (off + bytes + 255) & ~(size_t)255;
    return p;
  };
  _Float16* xn  = (_Float16*)alloc((size_t)MROWS * DDIM * 2);
  _Float16* Wqp = (_Float16*)alloc((size_t)DDIM * DDIM * 2);
  _Float16* Wkp = (_Float16*)alloc((size_t)DDIM * DDIM * 2);
  _Float16* Wvp = (_Float16*)alloc((size_t)DDIM * DDIM * 2);
  _Float16* Wop = (_Float16*)alloc((size_t)DDIM * DDIM * 2);
  _Float16* Qh  = (_Float16*)alloc((size_t)MROWS * DDIM * 2);
  _Float16* Kh  = (_Float16*)alloc((size_t)MROWS * DDIM * 2);
  _Float16* Vh  = (_Float16*)alloc((size_t)MROWS * DDIM * 2);
  _Float16* Oh  = (_Float16*)alloc((size_t)MROWS * DDIM * 2);
  float* rq   = (float*)alloc((size_t)2 * NHEAD * NGRP * 64 * 4);
  float* rk   = (float*)alloc((size_t)2 * NHEAD * NGRP * 64 * 4);
  float* mv   = (float*)alloc((size_t)2 * NHEAD * NGRP * 64 * 4);
  float* muw  = (float*)alloc((size_t)2 * NHEAD * NGRP * 4);
  float* intr = (float*)alloc((size_t)2 * NHEAD * NGRP * 64 * 4);

  dim3 blk(256);
  // 1. LayerNorm -> xn (f16)
  ln_kernel<<<MROWS / 8, blk, 0, stream>>>(x, gamma, beta, xn);
  // 2. Pre-pack weights into WMMA B-fragment layout
  prepack<<<1024, blk, 0, stream>>>(Wq, Wqp);
  prepack<<<1024, blk, 0, stream>>>(Wk, Wkp);
  prepack<<<1024, blk, 0, stream>>>(Wv, Wvp);
  prepack<<<1024, blk, 0, stream>>>(Wo, Wop);
  // 3. Q/K/V projections (WMMA)
  dim3 ggrid(MROWS / 128, DDIM / 64);
  gemm512<false><<<ggrid, blk, 0, stream>>>(xn, Wqp, nullptr, Qh, nullptr);
  gemm512<false><<<ggrid, blk, 0, stream>>>(xn, Wkp, nullptr, Kh, nullptr);
  gemm512<false><<<ggrid, blk, 0, stream>>>(xn, Wvp, nullptr, Vh, nullptr);
  // 4. L2 norm * sqrt(hd) on Q, K
  qknorm<<<(MROWS * NHEAD) / 8, blk, 0, stream>>>(Qh);
  qknorm<<<(MROWS * NHEAD) / 8, blk, 0, stream>>>(Kh);
  // 5. Group means
  group_stats<<<2 * NHEAD * NGRP, blk, 0, stream>>>(Qh, Kh, Vh, rq, rk, mv);
  // 6. Sibling weights + inter vectors
  sibling<<<2 * NHEAD, blk, 0, stream>>>(rq, rk, mv, muw, intr);
  // 7. Intra attention + combine (WMMA, async V staging)
  intra_attn<<<2 * NHEAD * NGRP, blk, 0, stream>>>(Qh, Kh, Vh, muw, intr, Oh);
  // 8. Output projection + residual (WMMA) -> d_out (f32)
  gemm512<true><<<ggrid, blk, 0, stream>>>(Oh, Wop, x, nullptr, (float*)d_out);
}
